// GemNetTDecoder_24163486008151
// MI455X (gfx1250) — compile-verified, matrix operand order checked by reading, not run on hardware
//
#include <hip/hip_runtime.h>
#include <hip/hip_bf16.h>

#define N_CRYST 256
#define ATOMS_PER 64
#define N_ATOMS (N_CRYST * ATOMS_PER)      // 16384
#define N_EDGES (N_ATOMS * 20)             // 327680
#define HID 128
#define LAT 256
#define NRAD 128
#define NBLOCK 3
#define MAXZ 100
#define CUTOFF 6.0f
#define NPAD 112                           // W_atom cols padded to 7*16

typedef __attribute__((ext_vector_type(16))) _Float16 v16h;
typedef __attribute__((ext_vector_type(8)))  float    v8f;
typedef __attribute__((ext_vector_type(4)))  int      v4i;

// ---------------------------------------------------------------------------
// CDNA5 async global->LDS copy (ASYNCcnt path), guarded so it always compiles
// Builtin signature (from compiler diagnostic): param0 = AS(1) int4*,
// param1 = AS(3) int4*, then imm offset + imm cpol.
// ---------------------------------------------------------------------------
#if defined(__HIP_DEVICE_COMPILE__) && __has_builtin(__builtin_amdgcn_global_load_async_to_lds_b128)
#define HAS_ASYNC_LDS 1
typedef __attribute__((address_space(1))) v4i* as1_v4i;
typedef __attribute__((address_space(3))) v4i* as3_v4i;
#else
#define HAS_ASYNC_LDS 0
#endif

// copy 16 bytes global -> LDS
__device__ __forceinline__ void cp16(_Float16* sDst, const _Float16* gSrc) {
#if HAS_ASYNC_LDS
    __builtin_amdgcn_global_load_async_to_lds_b128((as1_v4i)gSrc, (as3_v4i)sDst, 0, 0);
#else
    *(float4*)sDst = *(const float4*)gSrc;
#endif
}

__device__ __forceinline__ void cp_wait() {
#if HAS_ASYNC_LDS
#if __has_builtin(__builtin_amdgcn_s_wait_asynccnt)
    __builtin_amdgcn_s_wait_asynccnt(0);
#else
    asm volatile("s_wait_asynccnt 0" ::: "memory");
#endif
#endif
}

// ---------------------------------------------------------------------------
// WMMA helpers (CDNA5 wave32, V_WMMA_F32_16X16X32_F16)
// ---------------------------------------------------------------------------
__device__ __forceinline__ v8f wmma16(v16h a, v16h b, v8f c) {
    return __builtin_amdgcn_wmma_f32_16x16x32_f16(
        false, a, false, b, (short)0, c, false, false);
}

// A fragment: 16x32 f16 tile from LDS (row-major, row stride = ld halves).
// lanes 0-15 -> M=lane, kb=0; lanes 16-31 -> M=lane-16, kb=8;
// halves: pairs j<4 -> K=kb+2j,2j+1 ; j>=4 -> K=kb+16+2(j-4),+1.
__device__ __forceinline__ v16h load_A(const _Float16* sA, int ld, int k0, int lane) {
    int m  = lane & 15;
    int kb = (lane >> 4) * 8;
    const _Float16* row = sA + m * ld + k0 + kb;
    v16h a;
#pragma unroll
    for (int j = 0; j < 8; ++j) {
        int k = (j < 4) ? (2 * j) : (16 + 2 * (j - 4));
        a[2 * j]     = row[k];
        a[2 * j + 1] = row[k + 1];
    }
    return a;
}

// B fragment from pre-swizzled weights: tile (kt,nt) -> 32 lanes x 16 halves
// contiguous, i.e. one 32-byte per-lane load (2x global_load_b128).
__device__ __forceinline__ v16h load_Bp(const v16h* __restrict__ Wp, int NT,
                                        int kt, int nt, int lane) {
    return Wp[((kt * NT + nt) << 5) + lane];
}

template <int K>
__device__ __forceinline__ v8f gemm_tile(const _Float16* sA, const v16h* __restrict__ Wp,
                                         int NT, int nt, int lane) {
    v8f acc = {};
#pragma unroll
    for (int kt = 0; kt < K / 32; ++kt) {
        v16h a = load_A(sA, K, kt * 32, lane);
        v16h b = load_Bp(Wp, NT, kt, nt, lane);
        acc = wmma16(a, b, acc);
    }
    return acc;
}

__device__ __forceinline__ float silu(float x) {
    return x / (1.0f + __expf(-x));
}

// ---------------------------------------------------------------------------
// Weight pre-swizzle: W[K][srcN] f32 row-major -> fragment-native f16 layout
// Wp[((kt*NT+nt)*32 + lane)*16 + i], N padded with zeros up to NT*16.
// ---------------------------------------------------------------------------
__global__ void k_pack_w(const float* __restrict__ W, _Float16* __restrict__ Wp,
                         int K, int N, int srcN) {
    int idx = blockIdx.x * blockDim.x + threadIdx.x;   // over K*N
    if (idx >= K * N) return;
    int i    = idx & 15;
    int lane = (idx >> 4) & 31;
    int t    = idx >> 9;                               // kt*NT + nt
    int NT   = N >> 4;
    int kt   = t / NT, nt = t - kt * NT;
    int n = nt * 16 + (lane & 15);
    int k = kt * 32 + (lane >> 4) * 8 + ((i >= 8) ? 16 : 0) + (i & 7);
    Wp[idx] = (n < srcN) ? (_Float16)W[k * srcN + n] : (_Float16)0.0f;
}

__global__ void k_zero(float* __restrict__ p, int n) {
    int i = blockIdx.x * blockDim.x + threadIdx.x;
    if (i < n) p[i] = 0.0f;
}

// ---------------------------------------------------------------------------
// Geometry
// ---------------------------------------------------------------------------
__global__ void k_cart(const float* __restrict__ frac, const float* __restrict__ lengths,
                       const float* __restrict__ angles, float* __restrict__ cart) {
    int g = blockIdx.x * blockDim.x + threadIdx.x;
    if (g >= N_ATOMS) return;
    int cs = g >> 6;
    float a = lengths[cs * 3 + 0], b = lengths[cs * 3 + 1], c = lengths[cs * 3 + 2];
    const float D2R = 0.017453292519943295f;
    float al = angles[cs * 3 + 0] * D2R, be = angles[cs * 3 + 1] * D2R, ga = angles[cs * 3 + 2] * D2R;
    float ca = __cosf(al), cb = __cosf(be), cg = __cosf(ga);
    float sa = __sinf(al), sb = __sinf(be);
    float val = fminf(fmaxf((ca * cb - cg) / (sa * sb), -1.0f), 1.0f);
    float gs = acosf(val);
    float va0 = a * sb, va2 = a * cb;
    float vb0 = -b * sa * __cosf(gs), vb1 = b * sa * __sinf(gs), vb2 = b * ca;
    float f0 = frac[g * 3 + 0], f1 = frac[g * 3 + 1], f2 = frac[g * 3 + 2];
    cart[g * 3 + 0] = f0 * va0 + f1 * vb0;
    cart[g * 3 + 1] = f1 * vb1;
    cart[g * 3 + 2] = f0 * va2 + f1 * vb2 + f2 * c;
}

__global__ void k_geom(const int* __restrict__ ei, const float* __restrict__ cart,
                       float* __restrict__ dbuf, float* __restrict__ unit) {
    int e = blockIdx.x * blockDim.x + threadIdx.x;
    if (e >= N_EDGES) return;
    int s = ei[e], t = ei[N_EDGES + e];
    float vx = cart[t * 3 + 0] - cart[s * 3 + 0];
    float vy = cart[t * 3 + 1] - cart[s * 3 + 1];
    float vz = cart[t * 3 + 2] - cart[s * 3 + 2];
    float d = sqrtf(vx * vx + vy * vy + vz * vz) + 1e-8f;
    dbuf[e] = d;
    unit[e * 3 + 0] = vx / d;
    unit[e * 3 + 1] = vy / d;
    unit[e * 3 + 2] = vz / d;
}

// ---------------------------------------------------------------------------
// h = silu([atom_emb[type], z[batch]] @ W_in + b_in)   M=16384,K=384,N=128
// (staging needs f32->f16 conversion, VALU path)
// ---------------------------------------------------------------------------
__global__ void __launch_bounds__(256)
k_h(const int* __restrict__ types, const float* __restrict__ atom_emb,
    const float* __restrict__ z, const _Float16* __restrict__ Wp,
    const float* __restrict__ bias, float* __restrict__ h32, _Float16* __restrict__ h16) {
    __shared__ _Float16 sA[16 * 384];
    int m0 = blockIdx.x * 16;
    __builtin_prefetch(Wp, 0, 0);
    for (int idx = threadIdx.x; idx < 16 * 384; idx += 256) {
        int r = idx / 384, c = idx - r * 384;
        int g = m0 + r;
        float v = (c < HID) ? atom_emb[types[g] * HID + c]
                            : z[(g >> 6) * LAT + (c - HID)];
        sA[idx] = (_Float16)v;
    }
    __syncthreads();
    int lane = threadIdx.x & 31;
    int nt = threadIdx.x >> 5;
    v8f acc = gemm_tile<384>(sA, (const v16h*)Wp, 8, nt, lane);
    int n = nt * 16 + (lane & 15);
    float bn = bias[n];
#pragma unroll
    for (int r = 0; r < 8; ++r) {
        int m = (lane >> 4) * 8 + r;
        float s = silu(acc[r] + bn);
        h32[(size_t)(m0 + m) * HID + n] = s;
        h16[(size_t)(m0 + m) * HID + n] = (_Float16)s;
    }
}

// ---------------------------------------------------------------------------
// rbf_h = rbf(d) @ W_rbf  (basis generated on the fly)  M=E,K=128,N=128
// ---------------------------------------------------------------------------
__global__ void __launch_bounds__(256)
k_rbf(const float* __restrict__ dbuf, const _Float16* __restrict__ Wp,
      _Float16* __restrict__ rbf16) {
    __shared__ _Float16 sA[16 * NRAD];
    int e0 = blockIdx.x * 16;
    __builtin_prefetch(Wp, 0, 0);
    const float width = CUTOFF / NRAD;
    const float inv2w2 = 1.0f / (2.0f * width * width);
    const float cstep = CUTOFF / (NRAD - 1);
    for (int idx = threadIdx.x; idx < 16 * NRAD; idx += 256) {
        int r = idx >> 7, k = idx & 127;
        float d = dbuf[e0 + r];
        float env = 0.5f * (__cosf(3.14159265358979f * fminf(d / CUTOFF, 1.0f)) + 1.0f);
        float dk = d - cstep * k;
        sA[idx] = (_Float16)(__expf(-dk * dk * inv2w2) * env);
    }
    __syncthreads();
    int lane = threadIdx.x & 31;
    int nt = threadIdx.x >> 5;
    v8f acc = gemm_tile<NRAD>(sA, (const v16h*)Wp, 8, nt, lane);
    int n = nt * 16 + (lane & 15);
#pragma unroll
    for (int r = 0; r < 8; ++r) {
        int m = (lane >> 4) * 8 + r;
        rbf16[(size_t)(e0 + m) * HID + n] = (_Float16)acc[r];
    }
}

// ---------------------------------------------------------------------------
// m = silu([h[src], h[dst], rbf_h] @ W_edge + b_edge)   M=E,K=384,N=128
// Gathered rows staged with async global->LDS 16B chunks.
// ---------------------------------------------------------------------------
__global__ void __launch_bounds__(256)
k_minit(const int* __restrict__ ei, const _Float16* __restrict__ h16,
        const _Float16* __restrict__ rbf16, const _Float16* __restrict__ Wp,
        const float* __restrict__ bias, _Float16* __restrict__ m16) {
    __shared__ _Float16 sA[16 * 384];
    int e0 = blockIdx.x * 16;
    __builtin_prefetch(Wp, 0, 0);
    // 16 rows x 48 16-byte chunks = 768 chunks; 256 threads x 3
#pragma unroll
    for (int c = threadIdx.x; c < 768; c += 256) {
        int r = c / 48, j = c - r * 48;
        int e = e0 + r;
        const _Float16* src;
        if (j < 16)      src = h16 + (size_t)ei[e] * HID + j * 8;
        else if (j < 32) src = h16 + (size_t)ei[N_EDGES + e] * HID + (j - 16) * 8;
        else             src = rbf16 + (size_t)e * HID + (j - 32) * 8;
        cp16(&sA[r * 384 + j * 8], src);
    }
    cp_wait();
    __syncthreads();
    int lane = threadIdx.x & 31;
    int nt = threadIdx.x >> 5;
    v8f acc = gemm_tile<384>(sA, (const v16h*)Wp, 8, nt, lane);
    int n = nt * 16 + (lane & 15);
    float bn = bias[n];
#pragma unroll
    for (int r = 0; r < 8; ++r) {
        int m = (lane >> 4) * 8 + r;
        m16[(size_t)(e0 + m) * HID + n] = (_Float16)silu(acc[r] + bn);
    }
}

// ---------------------------------------------------------------------------
// One message block, fused: m2 = silu(m@W_msg+b) * (rbf_h@W_gate);
// agg[dst] += m2 (f32 atomics); m += m2.  Tiles staged via async->LDS.
// ---------------------------------------------------------------------------
__global__ void __launch_bounds__(256)
k_block(const int* __restrict__ ei, _Float16* __restrict__ m16,
        const _Float16* __restrict__ rbf16, const _Float16* __restrict__ Wm,
        const _Float16* __restrict__ Wg, const float* __restrict__ bm,
        float* __restrict__ agg) {
    __shared__ _Float16 sM[16 * HID];
    __shared__ _Float16 sR[16 * HID];
    __shared__ int sDst[16];
    int e0 = blockIdx.x * 16;
    __builtin_prefetch(Wm, 0, 0);
    __builtin_prefetch(Wg, 0, 0);
    // whole 16x128 f16 tile is one contiguous 4KB region: 256 x 16B chunks
    cp16(&sM[threadIdx.x * 8], m16 + (size_t)e0 * HID + threadIdx.x * 8);
    cp16(&sR[threadIdx.x * 8], rbf16 + (size_t)e0 * HID + threadIdx.x * 8);
    if (threadIdx.x < 16) sDst[threadIdx.x] = ei[N_EDGES + e0 + threadIdx.x];
    cp_wait();
    __syncthreads();
    int lane = threadIdx.x & 31;
    int nt = threadIdx.x >> 5;
    v8f accM = gemm_tile<HID>(sM, (const v16h*)Wm, 8, nt, lane);
    v8f accG = gemm_tile<HID>(sR, (const v16h*)Wg, 8, nt, lane);
    int n = nt * 16 + (lane & 15);
    float bn = bm[n];
#pragma unroll
    for (int r = 0; r < 8; ++r) {
        int m = (lane >> 4) * 8 + r;
        float m2 = silu(accM[r] + bn) * accG[r];
        atomicAdd(&agg[(size_t)sDst[m] * HID + n], m2);
        float mnew = (float)sM[m * HID + n] + m2;
        m16[(size_t)(e0 + m) * HID + n] = (_Float16)mnew;
    }
}

// ---------------------------------------------------------------------------
// h = h + silu(agg @ W_upd + b_upd)    M=16384,K=128,N=128  (f32->f16 staging)
// ---------------------------------------------------------------------------
__global__ void __launch_bounds__(256)
k_upd(const float* __restrict__ agg, const _Float16* __restrict__ Wp,
      const float* __restrict__ bias, float* __restrict__ h32, _Float16* __restrict__ h16) {
    __shared__ _Float16 sA[16 * HID];
    int m0 = blockIdx.x * 16;
    __builtin_prefetch(Wp, 0, 0);
    for (int idx = threadIdx.x; idx < 16 * HID; idx += 256) {
        int r = idx >> 7, c = idx & 127;
        sA[idx] = (_Float16)agg[(size_t)(m0 + r) * HID + c];
    }
    __syncthreads();
    int lane = threadIdx.x & 31;
    int nt = threadIdx.x >> 5;
    v8f acc = gemm_tile<HID>(sA, (const v16h*)Wp, 8, nt, lane);
    int n = nt * 16 + (lane & 15);
    float bn = bias[n];
#pragma unroll
    for (int r = 0; r < 8; ++r) {
        int m = (lane >> 4) * 8 + r;
        size_t o = (size_t)(m0 + m) * HID + n;
        float hv = h32[o] + silu(acc[r] + bn);
        h32[o] = hv;
        h16[o] = (_Float16)hv;
    }
}

// ---------------------------------------------------------------------------
// f_edge = (m . W_force) * unit ; scatter-add into out_diff[dst]
// ---------------------------------------------------------------------------
__global__ void k_force(const int* __restrict__ ei, const _Float16* __restrict__ m16,
                        const float* __restrict__ W_force, const float* __restrict__ unit,
                        float* __restrict__ out_diff) {
    int e = blockIdx.x * blockDim.x + threadIdx.x;
    if (e >= N_EDGES) return;
    const _Float16* row = m16 + (size_t)e * HID;
    float dot = 0.0f;
#pragma unroll 16
    for (int k = 0; k < HID; ++k) dot += (float)row[k] * W_force[k];
    int t = ei[N_EDGES + e];
#pragma unroll
    for (int j = 0; j < 3; ++j)
        atomicAdd(&out_diff[(size_t)t * 3 + j], dot * unit[e * 3 + j]);
}

// ---------------------------------------------------------------------------
// pred_atom_types = h @ W_atom + b_atom   M=16384,K=128,N=100 (padded 112)
// ---------------------------------------------------------------------------
__global__ void __launch_bounds__(224)
k_atom(const _Float16* __restrict__ h16, const _Float16* __restrict__ Wp,
       const float* __restrict__ bias, float* __restrict__ out_types) {
    __shared__ _Float16 sA[16 * HID];
    int m0 = blockIdx.x * 16;
    __builtin_prefetch(Wp, 0, 0);
    for (int c = threadIdx.x; c < 256; c += 224)   // 256 x 16B contiguous chunks
        cp16(&sA[c * 8], h16 + (size_t)m0 * HID + c * 8);
    cp_wait();
    __syncthreads();
    int lane = threadIdx.x & 31;
    int nt = threadIdx.x >> 5;
    v8f acc = gemm_tile<HID>(sA, (const v16h*)Wp, 7, nt, lane);
    int n = nt * 16 + (lane & 15);
    if (n < MAXZ) {
        float bn = bias[n];
#pragma unroll
        for (int r = 0; r < 8; ++r) {
            int m = (lane >> 4) * 8 + r;
            out_types[(size_t)(m0 + m) * MAXZ + n] = acc[r] + bn;
        }
    }
}

// ---------------------------------------------------------------------------
// Host launch
// ---------------------------------------------------------------------------
extern "C" void kernel_launch(void* const* d_in, const int* in_sizes, int n_in,
                              void* d_out, int out_size, void* d_ws, size_t ws_size,
                              hipStream_t stream) {
    (void)in_sizes; (void)n_in; (void)out_size; (void)ws_size;
    const float* z        = (const float*)d_in[0];
    const float* frac     = (const float*)d_in[1];
    const int*   types    = (const int*)d_in[2];
    const float* lengths  = (const float*)d_in[4];
    const float* angles   = (const float*)d_in[5];
    const int*   ei       = (const int*)d_in[6];
    const float* atom_emb = (const float*)d_in[7];
    const float* W_in     = (const float*)d_in[8];
    const float* b_in     = (const float*)d_in[9];
    const float* W_rbf    = (const float*)d_in[10];
    const float* W_edge   = (const float*)d_in[11];
    const float* b_edge   = (const float*)d_in[12];
    const float* W_msg    = (const float*)d_in[13];
    const float* b_msg    = (const float*)d_in[14];
    const float* W_gate   = (const float*)d_in[15];
    const float* W_upd    = (const float*)d_in[16];
    const float* b_upd    = (const float*)d_in[17];
    const float* W_force  = (const float*)d_in[18];
    const float* W_atom   = (const float*)d_in[19];
    const float* b_atom   = (const float*)d_in[20];

    float* out_diff  = (float*)d_out;                        // [N_ATOMS,3]
    float* out_types = (float*)d_out + (size_t)N_ATOMS * 3;  // [N_ATOMS,100]

    char* ws = (char*)d_ws;
    size_t off = 0;
    auto alloc = [&](size_t bytes) -> char* {
        char* p = ws + off;
        off = (off + bytes + 255) & ~(size_t)255;
        return p;
    };
    float*    cart   = (float*)   alloc((size_t)N_ATOMS * 3 * 4);
    float*    dbuf   = (float*)   alloc((size_t)N_EDGES * 4);
    float*    unit   = (float*)   alloc((size_t)N_EDGES * 3 * 4);
    float*    h32    = (float*)   alloc((size_t)N_ATOMS * HID * 4);
    _Float16* h16    = (_Float16*)alloc((size_t)N_ATOMS * HID * 2);
    _Float16* rbf16  = (_Float16*)alloc((size_t)N_EDGES * HID * 2);
    _Float16* m16    = (_Float16*)alloc((size_t)N_EDGES * HID * 2);
    float*    agg    = (float*)   alloc((size_t)N_ATOMS * HID * 4);
    _Float16* W_in16   = (_Float16*)alloc((size_t)(HID + LAT) * HID * 2);
    _Float16* W_rbf16  = (_Float16*)alloc((size_t)NRAD * HID * 2);
    _Float16* W_edge16 = (_Float16*)alloc((size_t)3 * HID * HID * 2);
    _Float16* W_msg16  = (_Float16*)alloc((size_t)NBLOCK * HID * HID * 2);
    _Float16* W_gate16 = (_Float16*)alloc((size_t)NBLOCK * HID * HID * 2);
    _Float16* W_upd16  = (_Float16*)alloc((size_t)NBLOCK * HID * HID * 2);
    _Float16* W_atom16 = (_Float16*)alloc((size_t)HID * NPAD * 2);

    auto pack = [&](const float* s, _Float16* d, int K, int N, int srcN) {
        int n = K * N;
        k_pack_w<<<(n + 255) / 256, 256, 0, stream>>>(s, d, K, N, srcN);
    };
    pack(W_in,   W_in16,   HID + LAT, HID, HID);
    pack(W_rbf,  W_rbf16,  NRAD, HID, HID);
    pack(W_edge, W_edge16, 3 * HID, HID, HID);
    for (int i = 0; i < NBLOCK; ++i) {
        pack(W_msg  + (size_t)i * HID * HID, W_msg16  + (size_t)i * HID * HID, HID, HID, HID);
        pack(W_gate + (size_t)i * HID * HID, W_gate16 + (size_t)i * HID * HID, HID, HID, HID);
        pack(W_upd  + (size_t)i * HID * HID, W_upd16  + (size_t)i * HID * HID, HID, HID, HID);
    }
    pack(W_atom, W_atom16, HID, NPAD, MAXZ);

    k_cart<<<(N_ATOMS + 255) / 256, 256, 0, stream>>>(frac, lengths, angles, cart);
    k_geom<<<(N_EDGES + 255) / 256, 256, 0, stream>>>(ei, cart, dbuf, unit);

    k_h<<<N_ATOMS / 16, 256, 0, stream>>>(types, atom_emb, z, W_in16, b_in, h32, h16);
    k_rbf<<<N_EDGES / 16, 256, 0, stream>>>(dbuf, W_rbf16, rbf16);
    k_minit<<<N_EDGES / 16, 256, 0, stream>>>(ei, h16, rbf16, W_edge16, b_edge, m16);

    for (int i = 0; i < NBLOCK; ++i) {
        int nagg = N_ATOMS * HID;
        k_zero<<<(nagg + 255) / 256, 256, 0, stream>>>(agg, nagg);
        k_block<<<N_EDGES / 16, 256, 0, stream>>>(
            ei, m16, rbf16,
            W_msg16 + (size_t)i * HID * HID,
            W_gate16 + (size_t)i * HID * HID,
            b_msg + (size_t)i * HID, agg);
        k_upd<<<N_ATOMS / 16, 256, 0, stream>>>(
            agg, W_upd16 + (size_t)i * HID * HID, b_upd + (size_t)i * HID, h32, h16);
    }

    k_zero<<<(N_ATOMS * 3 + 255) / 256, 256, 0, stream>>>(out_diff, N_ATOMS * 3);
    k_force<<<(N_EDGES + 255) / 256, 256, 0, stream>>>(ei, m16, W_force, unit, out_diff);
    k_atom<<<N_ATOMS / 16, 224, 0, stream>>>(h16, W_atom16, b_atom, out_types);
}
